// DSTFPE_40450001994246
// MI455X (gfx1250) — compile-verified, matrix operand order checked by reading, not run on hardware
//
#include <hip/hip_runtime.h>

typedef __attribute__((ext_vector_type(16))) _Float16 v16h;
typedef __attribute__((ext_vector_type(8)))  _Float16 v8h;
typedef __attribute__((ext_vector_type(8)))  float    v8f;

#define EDIM   512
#define HHEADS 8
#define DDIM   64
#define LNUM   3
#define RKDIM  256
#define XDIM   1024
#define OUTDIM 15
#define BNUM   8
#define NSEQ   2048
#define MTOK   (BNUM * NSEQ)

#define NBLK   64       // N tile per block (4 x 16 per wave)
#define MBLK   128      // M tile per block (8 waves x 16 rows)
#define BS_STR 40       // padded LDS row stride in halfs (80B, 16B-aligned)

__device__ __forceinline__ float relu6f(float x) { return fminf(fmaxf(x, 0.f), 6.f); }

// ---------------------------------------------------------------------------
// Batched WMMA GEMM:  C = act(A[M,K] @ W[K,N] + bias [+ Cadd])
// block = 256 threads (8 waves). Block tile = 128(M) x 64(N).
// B panel (32K x 64N) is staged once per block in LDS as f16 in
// fragment-ready [n][k] order; each wave computes a 16x64 strip: the A
// fragment is loaded once per K-step, all four B fragments are fetched from
// LDS up front, then four v_wmma issue back-to-back (DS latency hidden by
// WMMA/VALU co-execution).
// Requires: M % 128 == 0, N % 64 == 0, K % 32 == 0 (true for all call sites).
// ---------------------------------------------------------------------------
__global__ __launch_bounds__(256)
void wmma_gemm_kernel(const float* __restrict__ A, long lda, long sAb, long sAh,
                      const float* __restrict__ W, long ldw, long sWb, long sWh,
                      const float* __restrict__ bias,
                      const float* __restrict__ Cadd,
                      float* __restrict__ C, long ldc, long sCb, long sCh,
                      int M, int N, int K, int act, int Hdim)
{
    __shared__ _Float16 Bs[NBLK][BS_STR];              // 5120 B

    const int z  = blockIdx.z;
    const int zb = z / Hdim, zh = z % Hdim;
    A += zb * sAb + zh * sAh;
    W += zb * sWb + zh * sWh;
    C += zb * sCb + zh * sCh;
    const float* Cin = Cadd ? (Cadd + zb * sCb + zh * sCh) : nullptr;

    const int tid  = threadIdx.x;
    const int lane = tid & 31;
    const int wave = tid >> 5;
    const int half = lane >> 4;
    const int li   = lane & 15;
    const int n0   = blockIdx.x * NBLK;
    const int m0   = blockIdx.y * MBLK + wave * 16;

    // LDS-fill assignment: thread -> (n_local = tid&63, k-group = tid>>6)
    const int fn = tid & 63;
    const int fk = tid >> 6;                            // 0..3, covers k in [8*fk, 8*fk+8)

    const float* Arow = A + (size_t)(m0 + li) * lda;

    v8f acc[4] = {};

    for (int k0 = 0; k0 < K; k0 += 32) {
        // ---- stage B panel: Bs[n][k] = (f16) W[k0+k][n0+n], coalesced in n ----
        __syncthreads();
        {
            const float* wp = W + (size_t)(k0 + fk * 8) * ldw + n0 + fn;
            float f[8];
#pragma unroll
            for (int j = 0; j < 8; ++j) f[j] = wp[(size_t)j * ldw];
            v8h hv;
#pragma unroll
            for (int j = 0; j < 8; ++j) hv[j] = (_Float16)f[j];
            *(v8h*)(&Bs[fn][fk * 8]) = hv;
        }
        __syncthreads();

        // ---- A fragment (global, f32 -> f16) ----
        v16h a;
#pragma unroll
        for (int v = 0; v < 8; ++v) {
            const int kb = k0 + 2 * (v & 3) + 8 * half + 16 * (v >> 2);
            const float2 av = *(const float2*)(Arow + kb);
            a[2 * v]     = (_Float16)av.x;
            a[2 * v + 1] = (_Float16)av.y;
        }
        if (k0 + 32 < K) __builtin_prefetch(Arow + k0 + 32, 0, 1);

        // ---- fetch ALL four B fragments from LDS first ----
        v16h bfrag[4];
#pragma unroll
        for (int nt = 0; nt < 4; ++nt) {
            const _Float16* bp = &Bs[nt * 16 + li][0];
            const v8h lo = *(const v8h*)(bp + 8 * half);
            const v8h hi = *(const v8h*)(bp + 16 + 8 * half);
#pragma unroll
            for (int i = 0; i < 8; ++i) { bfrag[nt][i] = lo[i]; bfrag[nt][i + 8] = hi[i]; }
        }

        // ---- four WMMAs reusing the single A fragment ----
#pragma unroll
        for (int nt = 0; nt < 4; ++nt)
            acc[nt] = __builtin_amdgcn_wmma_f32_16x16x32_f16(false, a, false, bfrag[nt],
                                                             (short)0, acc[nt], false, false);
    }

    // ---- epilogue: bias + optional Cadd + activation ----
#pragma unroll
    for (int nt = 0; nt < 4; ++nt) {
        const int nc   = n0 + nt * 16 + li;
        const float bv = bias ? bias[nc] : 0.f;
#pragma unroll
        for (int r = 0; r < 8; ++r) {
            const int    m  = m0 + r + 8 * half;
            const size_t ci = (size_t)m * ldc + nc;
            float x = acc[nt][r] + bv;
            if (Cin) x += Cin[ci];
            if (act == 1)      x = relu6f(x);
            else if (act == 2) x = fmaxf(x, 0.f);
            C[ci] = x;
        }
    }
}

// ---------------------------------------------------------------------------
// kv[b,h] = sum_n k[b,n,h,:]^T v[b,n,h,:]   (64x64 per (b,h)), WMMA reduction
// grid.x = B*H, block = 256 (8 waves, each wave does 2 of the 16 16x16 tiles)
// ---------------------------------------------------------------------------
__global__ __launch_bounds__(256)
void kv_kernel(const float* __restrict__ kk, const float* __restrict__ vv,
               float* __restrict__ kv)
{
    const int bh = blockIdx.x;
    const int b  = bh >> 3, h = bh & 7;
    const int wave = threadIdx.x >> 5;
    const int lane = threadIdx.x & 31;
    const int half = lane >> 4, li = lane & 15;
    const float* kb = kk + (size_t)b * NSEQ * EDIM + h * DDIM;
    const float* vb = vv + (size_t)b * NSEQ * EDIM + h * DDIM;
    float* kvb = kv + (size_t)bh * DDIM * DDIM;

    for (int t = wave; t < 16; t += 8) {               // wave-uniform
        const int dA = (t >> 2) * 16;
        const int dB = (t & 3) * 16;
        v8f acc = {};
        for (int n0 = 0; n0 < NSEQ; n0 += 32) {
            v16h a, bfr;
#pragma unroll
            for (int v = 0; v < 8; ++v) {
                const int kn = n0 + 2 * (v & 3) + 8 * half + 16 * (v >> 2);
                a[2 * v]       = (_Float16)kb[(size_t)kn * EDIM + dA + li];
                a[2 * v + 1]   = (_Float16)kb[(size_t)(kn + 1) * EDIM + dA + li];
                bfr[2 * v]     = (_Float16)vb[(size_t)kn * EDIM + dB + li];
                bfr[2 * v + 1] = (_Float16)vb[(size_t)(kn + 1) * EDIM + dB + li];
            }
            acc = __builtin_amdgcn_wmma_f32_16x16x32_f16(false, a, false, bfr,
                                                         (short)0, acc, false, false);
        }
#pragma unroll
        for (int r = 0; r < 8; ++r)
            kvb[(dA + r + 8 * half) * DDIM + dB + li] = acc[r];
    }
}

// ---------------------------------------------------------------------------
// Elementwise / reduction helper kernels
// ---------------------------------------------------------------------------
__global__ __launch_bounds__(256)
void eluplus_kernel(float* __restrict__ x, int n)
{
    int i = blockIdx.x * 256 + threadIdx.x;
    if (i >= n) return;
    float v = x[i];
    x[i] = (v > 0.f) ? (v + 1.f) : __expf(v);          // elu(x)+1
}

__global__ __launch_bounds__(64)
void ksum_kernel(const float* __restrict__ k, float* __restrict__ ks)
{
    const int bh = blockIdx.x;                          // B*H
    const int b = bh >> 3, h = bh & 7, d = threadIdx.x; // d in [0,64)
    const float* kp = k + (size_t)b * NSEQ * EDIM + h * DDIM + d;
    float s = 0.f;
    for (int n = 0; n < NSEQ; ++n) s += kp[(size_t)n * EDIM];
    ks[bh * DDIM + d] = s;
}

__global__ __launch_bounds__(256)
void denom_kernel(const float* __restrict__ q, const float* __restrict__ ks,
                  float* __restrict__ den, int n)      // n = MTOK*H
{
    int i = blockIdx.x * 256 + threadIdx.x;
    if (i >= n) return;
    const int h  = i & 7;
    const int bn = i >> 3;
    const int b  = bn / NSEQ;
    const float* qp = q + (size_t)bn * EDIM + h * DDIM;
    const float* kp = ks + (b * HHEADS + h) * DDIM;
    float s = 0.f;
#pragma unroll
    for (int d = 0; d < DDIM; ++d) s += qp[d] * kp[d];
    den[i] = s;
}

__global__ __launch_bounds__(256)
void divide_kernel(float* __restrict__ a, const float* __restrict__ den, int n)
{
    int i = blockIdx.x * 256 + threadIdx.x;
    if (i >= n) return;
    const int tok = i >> 9;                             // /512
    const int h   = (i & 511) >> 6;                     // col/64
    a[i] = a[i] / (den[tok * HHEADS + h] + 1e-6f);
}

// y = act(LN(x + res)) over rows of 512; one wave per row
__global__ __launch_bounds__(256)
void ln_kernel(const float* __restrict__ x, const float* __restrict__ res,
               const float* __restrict__ g, const float* __restrict__ b,
               float* __restrict__ y, int rows, int act)
{
    const int wave = threadIdx.x >> 5, lane = threadIdx.x & 31;
    const int row = blockIdx.x * 8 + wave;
    if (row >= rows) return;
    const float* xr = x + (size_t)row * EDIM;
    const float* rr = res ? res + (size_t)row * EDIM : nullptr;
    float vals[16];
    float s = 0.f;
#pragma unroll
    for (int i = 0; i < 16; ++i) {
        float t = xr[lane + 32 * i];
        if (rr) t += rr[lane + 32 * i];
        vals[i] = t;
        s += t;
    }
#pragma unroll
    for (int m = 16; m >= 1; m >>= 1) s += __shfl_xor(s, m, 32);
    const float mean = s * (1.f / EDIM);
    float vs = 0.f;
#pragma unroll
    for (int i = 0; i < 16; ++i) { float d = vals[i] - mean; vs += d * d; }
#pragma unroll
    for (int m = 16; m >= 1; m >>= 1) vs += __shfl_xor(vs, m, 32);
    const float inv = rsqrtf(vs * (1.f / EDIM) + 1e-5f);
#pragma unroll
    for (int i = 0; i < 16; ++i) {
        const int c = lane + 32 * i;
        float o = (vals[i] - mean) * inv * g[c] + b[c];
        if (act == 2) o = fmaxf(o, 0.f);
        y[(size_t)row * EDIM + c] = o;
    }
}

// depthwise conv1d(k=3,pad=1) + BN(eval, mean0 var1) + relu6
__global__ __launch_bounds__(256)
void dwconv_kernel(const float* __restrict__ h1, const float* __restrict__ cw,
                   const float* __restrict__ cb, const float* __restrict__ bng,
                   const float* __restrict__ bnb, float* __restrict__ h2, int n)
{
    int i = blockIdx.x * 256 + threadIdx.x;
    if (i >= n) return;
    const int x = i & (XDIM - 1);
    const int s = (i >> 10) & (NSEQ - 1);
    const float a = (s > 0)        ? h1[i - XDIM] : 0.f;
    const float m = h1[i];
    const float c = (s < NSEQ - 1) ? h1[i + XDIM] : 0.f;
    float t = a * cw[x * 3 + 0] + m * cw[x * 3 + 1] + c * cw[x * 3 + 2] + cb[x];
    t = t * rsqrtf(1.f + 1e-5f) * bng[x] + bnb[x];
    h2[i] = relu6f(t);
}

// 2-way softmax gate + blend: one wave per token
__global__ __launch_bounds__(256)
void gate_combine_kernel(const float* __restrict__ gh, const float* __restrict__ gw2,
                         const float* __restrict__ gb2, const float* __restrict__ x1,
                         const float* __restrict__ c, float* __restrict__ out, int rows)
{
    const int wave = threadIdx.x >> 5, lane = threadIdx.x & 31;
    const int row = blockIdx.x * 8 + wave;
    if (row >= rows) return;
    const float* h = gh + (size_t)row * 128;
    float p0 = 0.f, p1 = 0.f;
#pragma unroll
    for (int i = lane; i < 128; i += 32) {
        const float hv = h[i];
        p0 += hv * gw2[i * 2 + 0];
        p1 += hv * gw2[i * 2 + 1];
    }
#pragma unroll
    for (int m = 16; m >= 1; m >>= 1) {
        p0 += __shfl_xor(p0, m, 32);
        p1 += __shfl_xor(p1, m, 32);
    }
    p0 += gb2[0]; p1 += gb2[1];
    const float mx = fmaxf(p0, p1);
    const float e0 = __expf(p0 - mx), e1 = __expf(p1 - mx);
    const float g0 = e0 / (e0 + e1), g1 = e1 / (e0 + e1);
    const size_t base = (size_t)row * EDIM;
#pragma unroll
    for (int i = lane; i < EDIM; i += 32)
        out[base + i] = x1[base + i] * g0 + c[base + i] * g1;
}

// final 64 -> 15 head (thread per output element)
__global__ __launch_bounds__(256)
void head15_kernel(const float* __restrict__ r2, const float* __restrict__ rw3,
                   const float* __restrict__ rb3, float* __restrict__ out, int n)
{
    int i = blockIdx.x * 256 + threadIdx.x;
    if (i >= n) return;
    const int t = i / OUTDIM, o = i % OUTDIM;
    const float* rp = r2 + (size_t)t * DDIM;
    float s = rb3[o];
#pragma unroll
    for (int d = 0; d < DDIM; ++d) s += rp[d] * rw3[d * OUTDIM + o];
    out[i] = s;
}

// ---------------------------------------------------------------------------
// Host-side helpers
// ---------------------------------------------------------------------------
static inline void launch_gemm(hipStream_t st, const float* A, long lda,
                               const float* W, long ldw, const float* bias,
                               const float* Cadd, float* C, long ldc,
                               int M, int N, int K, int act,
                               int batches = 1, int hdim = 1,
                               long sAb = 0, long sAh = 0, long sWb = 0,
                               long sWh = 0, long sCb = 0, long sCh = 0)
{
    dim3 grid(N / NBLK, M / MBLK, batches);
    wmma_gemm_kernel<<<grid, dim3(256), 0, st>>>(A, lda, sAb, sAh, W, ldw, sWb, sWh,
                                                 bias, Cadd, C, ldc, sCb, sCh,
                                                 M, N, K, act, hdim);
}

static void run_attention(hipStream_t st, const float* xq, const float* xkv,
                          const float* dwp, const float* uwp, const float* ubp,
                          const float* owp, const float* obp,
                          float* tR, float* q, float* k, float* v,
                          float* kvb, float* ksb, float* denb,
                          float* attn, float* outb)
{
    const int tot = MTOK * EDIM;
    // q/k/v: low-rank down+up projections
    launch_gemm(st, xq,  EDIM,  dwp + 0L * EDIM * RKDIM, RKDIM, nullptr, nullptr, tR, RKDIM, MTOK, RKDIM, EDIM, 0);
    launch_gemm(st, tR,  RKDIM, uwp + 0L * RKDIM * EDIM, EDIM,  ubp + 0 * EDIM, nullptr, q, EDIM, MTOK, EDIM, RKDIM, 0);
    launch_gemm(st, xkv, EDIM,  dwp + 1L * EDIM * RKDIM, RKDIM, nullptr, nullptr, tR, RKDIM, MTOK, RKDIM, EDIM, 0);
    launch_gemm(st, tR,  RKDIM, uwp + 1L * RKDIM * EDIM, EDIM,  ubp + 1 * EDIM, nullptr, k, EDIM, MTOK, EDIM, RKDIM, 0);
    launch_gemm(st, xkv, EDIM,  dwp + 2L * EDIM * RKDIM, RKDIM, nullptr, nullptr, tR, RKDIM, MTOK, RKDIM, EDIM, 0);
    launch_gemm(st, tR,  RKDIM, uwp + 2L * RKDIM * EDIM, EDIM,  ubp + 2 * EDIM, nullptr, v, EDIM, MTOK, EDIM, RKDIM, 0);

    eluplus_kernel<<<(tot + 255) / 256, 256, 0, st>>>(q, tot);
    eluplus_kernel<<<(tot + 255) / 256, 256, 0, st>>>(k, tot);
    ksum_kernel<<<BNUM * HHEADS, 64, 0, st>>>(k, ksb);
    kv_kernel<<<BNUM * HHEADS, 256, 0, st>>>(k, v, kvb);
    const int dn = MTOK * HHEADS;
    denom_kernel<<<(dn + 255) / 256, 256, 0, st>>>(q, ksb, denb, dn);

    // num = q @ kv, batched over (b,h)
    launch_gemm(st, q, EDIM, kvb, DDIM, nullptr, nullptr, attn, EDIM,
                NSEQ, DDIM, DDIM, 0, BNUM * HHEADS, HHEADS,
                (long)NSEQ * EDIM, DDIM,
                (long)HHEADS * DDIM * DDIM, (long)DDIM * DDIM,
                (long)NSEQ * EDIM, DDIM);
    divide_kernel<<<(tot + 255) / 256, 256, 0, st>>>(attn, denb, tot);
    // output projection
    launch_gemm(st, attn, EDIM, owp, EDIM, obp, nullptr, outb, EDIM, MTOK, EDIM, EDIM, 0);
}

extern "C" void kernel_launch(void* const* d_in, const int* in_sizes, int n_in,
                              void* d_out, int out_size, void* d_ws, size_t ws_size,
                              hipStream_t stream)
{
    const float* bodyIn = (const float*)d_in[0];
    const float* limbIn = (const float*)d_in[1];
    const float* dw  = (const float*)d_in[2];
    const float* uw  = (const float*)d_in[3];
    const float* ub  = (const float*)d_in[4];
    const float* ow  = (const float*)d_in[5];
    const float* ob  = (const float*)d_in[6];
    const float* w1  = (const float*)d_in[7];
    const float* b1  = (const float*)d_in[8];
    const float* cw  = (const float*)d_in[9];
    const float* cb  = (const float*)d_in[10];
    const float* bng = (const float*)d_in[11];
    const float* bnb = (const float*)d_in[12];
    const float* w2  = (const float*)d_in[13];
    const float* b2  = (const float*)d_in[14];
    const float* lng = (const float*)d_in[15];
    const float* lnb = (const float*)d_in[16];
    const float* gw1 = (const float*)d_in[17];
    const float* gb1 = (const float*)d_in[18];
    const float* gw2 = (const float*)d_in[19];
    const float* gb2 = (const float*)d_in[20];
    const float* fw1 = (const float*)d_in[21];
    const float* fb1 = (const float*)d_in[22];
    const float* fw2 = (const float*)d_in[23];
    const float* fb2 = (const float*)d_in[24];
    const float* flng = (const float*)d_in[25];
    const float* flnb = (const float*)d_in[26];
    const float* rw1 = (const float*)d_in[27];
    const float* rb1 = (const float*)d_in[28];
    const float* rw2 = (const float*)d_in[29];
    const float* rb2 = (const float*)d_in[30];
    const float* rw3 = (const float*)d_in[31];
    const float* rb3 = (const float*)d_in[32];
    float* out = (float*)d_out;

    const size_t ME = (size_t)MTOK * EDIM;              // 8.4M floats per region
    const size_t need = (12 * ME + 64 * 4096 + 4096 + (size_t)MTOK * HHEADS) * sizeof(float);
    if (ws_size < need) return;

    float* ws = (float*)d_ws;
    float* rg[12];
    for (int i = 0; i < 12; ++i) rg[i] = ws + i * ME;
    float* b_cur = rg[0];  float* l_cur = rg[1];
    float* b1b   = rg[2];  float* l1b   = rg[3];
    float* bcb   = rg[4];  float* lcb   = rg[5];
    float* qb    = rg[6];  float* kb    = rg[7];  float* vb = rg[8];
    float* tR    = rg[9];
    float* ghb   = rg[9] + (size_t)MTOK * RKDIM;        // reused after down-proj
    float* attnb = rg[10]; float* sbuf  = rg[11];
    float* kvb   = ws + 12 * ME;
    float* ksb   = kvb + 64 * 4096;
    float* denb  = ksb + 64 * 64;
    // aliases for later phases (all lifetime-disjoint)
    float* body2 = rg[6]; float* limb2 = rg[7];
    float* h1b   = rg[4];                               // spans rg[4..5] = M x 1024
    float* h2b   = rg[2];                               // spans rg[2..3] = M x 1024
    float* ffno  = rg[10];
    float* fusedb = rg[9]; float* fused2 = rg[2];
    float* r1b    = rg[10]; float* r2b   = rg[11];

    hipMemcpyAsync(b_cur, bodyIn, ME * sizeof(float), hipMemcpyDeviceToDevice, stream);
    hipMemcpyAsync(l_cur, limbIn, ME * sizeof(float), hipMemcpyDeviceToDevice, stream);

    const int totX = MTOK * XDIM;

    for (int l = 0; l < LNUM; ++l) {
        auto DWP = [&](int a) { return dw + ((size_t)(l * 4 + a) * 3) * EDIM * RKDIM; };
        auto UWP = [&](int a) { return uw + ((size_t)(l * 4 + a) * 3) * RKDIM * EDIM; };
        auto UBP = [&](int a) { return ub + ((size_t)(l * 4 + a) * 3) * EDIM; };
        auto OWP = [&](int a) { return ow + (size_t)(l * 4 + a) * EDIM * EDIM; };
        auto OBP = [&](int a) { return ob + (size_t)(l * 4 + a) * EDIM; };
        auto LNG = [&](int i) { return lng + (size_t)(l * 5 + i) * EDIM; };
        auto LNB = [&](int i) { return lnb + (size_t)(l * 5 + i) * EDIM; };

        // self attentions + post-LN
        run_attention(stream, b_cur, b_cur, DWP(0), UWP(0), UBP(0), OWP(0), OBP(0),
                      tR, qb, kb, vb, kvb, ksb, denb, attnb, sbuf);
        ln_kernel<<<MTOK / 8, 256, 0, stream>>>(sbuf, b_cur, LNG(0), LNB(0), b1b, MTOK, 0);
        run_attention(stream, l_cur, l_cur, DWP(1), UWP(1), UBP(1), OWP(1), OBP(1),
                      tR, qb, kb, vb, kvb, ksb, denb, attnb, sbuf);
        ln_kernel<<<MTOK / 8, 256, 0, stream>>>(sbuf, l_cur, LNG(1), LNB(1), l1b, MTOK, 0);

        // cross attentions
        run_attention(stream, b1b, l1b, DWP(2), UWP(2), UBP(2), OWP(2), OBP(2),
                      tR, qb, kb, vb, kvb, ksb, denb, attnb, bcb);
        run_attention(stream, l1b, b1b, DWP(3), UWP(3), UBP(3), OWP(3), OBP(3),
                      tR, qb, kb, vb, kvb, ksb, denb, attnb, lcb);

        // gating (concat realized as two accumulating GEMMs)
        const float* gw1l = gw1 + (size_t)l * 2 * EDIM * 128;
        const float* gb1l = gb1 + (size_t)l * 128;
        const float* gw2l = gw2 + (size_t)l * 128 * 2;
        const float* gb2l = gb2 + (size_t)l * 2;
        launch_gemm(stream, b1b, EDIM, gw1l, 128, nullptr, nullptr, ghb, 128, MTOK, 128, EDIM, 0);
        launch_gemm(stream, bcb, EDIM, gw1l + (size_t)EDIM * 128, 128, gb1l, ghb, ghb, 128, MTOK, 128, EDIM, 1);
        gate_combine_kernel<<<MTOK / 8, 256, 0, stream>>>(ghb, gw2l, gb2l, b1b, bcb, body2, MTOK);
        ln_kernel<<<MTOK / 8, 256, 0, stream>>>(body2, nullptr, LNG(2), LNB(2), body2, MTOK, 0);
        launch_gemm(stream, l1b, EDIM, gw1l, 128, nullptr, nullptr, ghb, 128, MTOK, 128, EDIM, 0);
        launch_gemm(stream, lcb, EDIM, gw1l + (size_t)EDIM * 128, 128, gb1l, ghb, ghb, 128, MTOK, 128, EDIM, 1);
        gate_combine_kernel<<<MTOK / 8, 256, 0, stream>>>(ghb, gw2l, gb2l, l1b, lcb, limb2, MTOK);
        // source bug preserved: limb uses norm2_body params too
        ln_kernel<<<MTOK / 8, 256, 0, stream>>>(limb2, nullptr, LNG(2), LNB(2), limb2, MTOK, 0);

        // FFN per stream
        for (int st2 = 0; st2 < 2; ++st2) {
            const float* w1l  = w1  + (size_t)(l * 2 + st2) * EDIM * XDIM;
            const float* b1l  = b1  + (size_t)(l * 2 + st2) * XDIM;
            const float* cwl  = cw  + (size_t)(l * 2 + st2) * XDIM * 3;
            const float* cbl  = cb  + (size_t)(l * 2 + st2) * XDIM;
            const float* bngl = bng + (size_t)(l * 2 + st2) * XDIM;
            const float* bnbl = bnb + (size_t)(l * 2 + st2) * XDIM;
            const float* w2l  = w2  + (size_t)(l * 2 + st2) * XDIM * EDIM;
            const float* b2l  = b2  + (size_t)(l * 2 + st2) * EDIM;
            float* xin  = st2 ? limb2 : body2;
            float* xout = st2 ? l_cur : b_cur;
            launch_gemm(stream, xin, EDIM, w1l, XDIM, b1l, nullptr, h1b, XDIM, MTOK, XDIM, EDIM, 1);
            dwconv_kernel<<<(totX + 255) / 256, 256, 0, stream>>>(h1b, cwl, cbl, bngl, bnbl, h2b, totX);
            launch_gemm(stream, h2b, XDIM, w2l, EDIM, b2l, nullptr, ffno, EDIM, MTOK, EDIM, XDIM, 0);
            ln_kernel<<<MTOK / 8, 256, 0, stream>>>(ffno, xin, LNG(3 + st2), LNB(3 + st2), xout, MTOK, 0);
        }
    }

    // fusion head
    launch_gemm(stream, b_cur, EDIM, fw1, 256, nullptr, nullptr, fusedb, 256, MTOK, 256, EDIM, 0);
    launch_gemm(stream, l_cur, EDIM, fw1 + (size_t)EDIM * 256, 256, fb1, fusedb, fusedb, 256, MTOK, 256, EDIM, 1);
    launch_gemm(stream, fusedb, 256, fw2, EDIM, fb2, nullptr, fused2, EDIM, MTOK, EDIM, 256, 0);
    ln_kernel<<<MTOK / 8, 256, 0, stream>>>(fused2, nullptr, flng, flnb, fused2, MTOK, 2);
    launch_gemm(stream, fused2, EDIM, rw1, 128, rb1, nullptr, r1b, 128, MTOK, 128, EDIM, 1);
    launch_gemm(stream, r1b, 128, rw2, 64, rb2, nullptr, r2b, 64, MTOK, 64, 128, 1);
    const int on = MTOK * OUTDIM;
    head15_kernel<<<(on + 255) / 256, 256, 0, stream>>>(r2b, rw3, rb3, out, on);

    (void)in_sizes; (void)n_in; (void)out_size;
}
